// MultiHeadConvQKAttention_80891414053140
// MI455X (gfx1250) — compile-verified, matrix operand order checked by reading, not run on hardware
//
#include <hip/hip_runtime.h>
#include <hip/hip_bf16.h>
#include <math.h>

// ---------------------------------------------------------------------------
// Fused MultiHeadConv(3x3 window)Attention block for MI455X (gfx1250, wave32)
//   x -> qkv (WMMA) -> local attention+softmax (VALU/LDS) -> +res, LN1
//     -> FFN 64->128->64 (WMMA) -> +res, LN2 -> fuse 1x1 conv (WMMA) -> BN+ReLU
// One workgroup = one 8x8 pixel tile (+1 halo). All intermediates in LDS.
// HBM traffic ~= read x once + write out once (~320 MB total @ 23.3 TB/s).
// Weight (B) fragments are hoisted per output-column and reused across all
// M-tiles; K/V LDS tiles padded to 112 rows so C-scatter is branch-free.
// ---------------------------------------------------------------------------

typedef _Float16 h16;
typedef __attribute__((ext_vector_type(16))) _Float16 v16h;
typedef __attribute__((ext_vector_type(8)))  float    v8f;

#define BATCH 4
#define IMG   384
#define CCH   64          // channels
#define FFD   128         // ffn hidden
#define TILE  8           // spatial tile (pixels per side)
#define HALO  10          // TILE + 2
#define NPIX  64          // TILE*TILE
#define NHALO 100         // HALO*HALO
#define KVROWS 112        // NHALO padded to 7 full 16-row M-tiles
#define LN_EPS 1e-3f
#define BN_EPS 1e-3f

// k index held by half `h` (0..15) of a v16h A/B fragment for lane-group kg
// (per CDNA5 ISA 7.12.2: vgpr v<4 -> K=2v,2v+1 ; v>=4 -> K=16+2(v-4),... ;
//  lanes 16-31 add +8)
__device__ __forceinline__ int wmma_k_of_half(int h, int kg) {
  int v  = h >> 1;
  int kb = (v < 4) ? (v << 1) : (16 + ((v - 4) << 1));
  return kb + (h & 1) + (kg << 3);
}

// A fragment: 16x32 f16; lane holds one matrix row (rowp), k varies per half.
__device__ __forceinline__ v16h pack_a_lds(const h16* __restrict__ rowp,
                                           int kofs, int kg) {
  v16h a;
#pragma unroll
  for (int h = 0; h < 16; ++h) a[h] = rowp[kofs + wmma_k_of_half(h, kg)];
  return a;
}

// B fragment: 32x16 f16; lane holds one output column n of row-major W[K][ldn].
// Weights are fp32 in global; 128 KB shared by all 9216 workgroups -> L2-hot.
__device__ __forceinline__ v16h pack_b_glb(const float* __restrict__ w,
                                           int ldn, int n, int kofs, int kg) {
  v16h b;
#pragma unroll
  for (int h = 0; h < 16; ++h)
    b[h] = (h16)w[(kofs + wmma_k_of_half(h, kg)) * ldn + n];
  return b;
}

__device__ __forceinline__ v8f wmma_f16(v16h a, v16h b, v8f c) {
  return __builtin_amdgcn_wmma_f32_16x16x32_f16(
      /*neg_a=*/false, a, /*neg_b=*/false, b,
      /*c_mod=*/(short)0, c, /*reuse_a=*/false, /*reuse_b=*/false);
}

__global__ __launch_bounds__(256, 2)
void mhca_fused_kernel(const float* __restrict__ x,
                       const float* __restrict__ wq,
                       const float* __restrict__ wk,
                       const float* __restrict__ wv,
                       const float* __restrict__ ln1_g,
                       const float* __restrict__ ln1_b,
                       const float* __restrict__ ff1_w,
                       const float* __restrict__ ff1_b,
                       const float* __restrict__ ff2_w,
                       const float* __restrict__ ff2_b,
                       const float* __restrict__ ln2_g,
                       const float* __restrict__ ln2_b,
                       const float* __restrict__ w_fuse,
                       const float* __restrict__ bn_g,
                       const float* __restrict__ bn_b,
                       const float* __restrict__ bn_mean,
                       const float* __restrict__ bn_var,
                       float* __restrict__ out) {
  // ---- LDS: 49664 bytes, phase-overlaid ------------------------------------
  // [0     ,  6400) XH : x halo tile, 100x64 f16  (ph0-2)  -> O2 64x64 (ph4+)
  // [6400  , 10496) QS : q, 64x64 f16             (ph1-2)  -> O1 (LN1 out)
  // [10496 , 17664) KS : k halo, 112x64 f16       (ph1-2)  -> T1 64x128 (ph3-4)
  // [17664 , 24832) VS : v halo, 112x64 f16       (ph1-2)     (T1 spans into VS)
  __shared__ h16 smem[24832];
  h16* XH = smem;                  // 6400
  h16* QS = smem + 6400;           // 4096  (aliased: O1)
  h16* KS = smem + 10496;          // 7168  (aliased: T1 = 8192 halves)
  h16* VS = smem + 17664;          // 7168
  h16* O1 = QS;
  h16* T1 = KS;
  h16* O2 = XH;

  const int tid   = threadIdx.x;
  const int lane  = tid & 31;
  const int wv_id = tid >> 5;      // wave id 0..7
  const int lm    = lane & 15;     // within-fragment row/col
  const int kg    = lane >> 4;     // lane half-group
  const int b     = blockIdx.z;
  const int ty0   = blockIdx.y * TILE;
  const int tx0   = blockIdx.x * TILE;

  // ===== Phase 0: stage x halo tile (10x10x64) into LDS as f16 ==============
  for (int idx = tid; idx < NHALO * (CCH / 4); idx += 256) {
    int hp = idx >> 4;                 // halo pixel 0..99
    int c4 = (idx & 15) << 2;          // channel group of 4
    int gy = ty0 + (hp / HALO) - 1;
    int gx = tx0 + (hp % HALO) - 1;
    float4 v = make_float4(0.f, 0.f, 0.f, 0.f);
    if (gy >= 0 && gy < IMG && gx >= 0 && gx < IMG)
      v = *(const float4*)&x[(((size_t)b * IMG + gy) * IMG + gx) * CCH + c4];
    h16* d = &XH[hp * CCH + c4];
    d[0] = (h16)v.x; d[1] = (h16)v.y; d[2] = (h16)v.z; d[3] = (h16)v.w;
  }
  __syncthreads();

  // ===== Phase 1: q/k/v projections via WMMA ================================
  // Column-major work split: col 0-3 -> q (4 Mtiles), 4-7 -> k (7 Mtiles),
  // 8-11 -> v (7 Mtiles). B fragments loaded once per column, reused across
  // all M-tiles. 144 WMMA / workgroup total.
  for (int col = wv_id; col < 12; col += 8) {
    int sel = col >> 2;                          // 0=q 1=k 2=v (wave-uniform)
    int nt  = col & 3;
    int n   = nt * 16 + lm;
    const float* W = (sel == 0) ? wq : (sel == 1) ? wk : wv;
    h16* dst = (sel == 0) ? QS : (sel == 1) ? KS : VS;
    v16h b0 = pack_b_glb(W, CCH, n, 0, kg);
    v16h b1 = pack_b_glb(W, CCH, n, 32, kg);
    int mtn = (sel == 0) ? 4 : 7;
    for (int mt = 0; mt < mtn; ++mt) {
      int gm = mt * 16 + lm;                     // output row for this lane
      int hp;                                    // source halo-pixel row
      if (sel == 0) hp = ((gm >> 3) + 1) * HALO + (gm & 7) + 1;  // interior
      else          hp = (gm < NHALO) ? gm : (NHALO - 1);        // clamp pad
      const h16* arow = &XH[hp * CCH];
      v8f c = {};
      c = wmma_f16(pack_a_lds(arow, 0, kg), b0, c);
      c = wmma_f16(pack_a_lds(arow, 32, kg), b1, c);
      // C scatter: vgpr r -> row r (lanes<16) / r+8 (lanes>=16), col lane%16.
      // KS/VS padded to 112 rows => no guard, straight ds_store_b16.
#pragma unroll
      for (int r = 0; r < 8; ++r)
        dst[(mt * 16 + r + (kg << 3)) * CCH + n] = (h16)c[r];
    }
  }
  __syncthreads();

  // ===== Phase 2: 3x3 window attention + residual + LayerNorm1 ==============
  // thread = (pixel p, head h): lanes 4p..4p+3 share a pixel -> shfl reduce.
  {
    int p  = tid >> 2;                 // 0..63
    int hd = tid & 3;                  // head
    int iy = p >> 3, ix = p & 7;
    int cb = hd << 4;                  // channel base (DEPTH=16)

    float qv[16];
#pragma unroll
    for (int j = 0; j < 16; ++j) qv[j] = (float)QS[p * CCH + cb + j];

    float sims[9], mx = -3.0e38f;
#pragma unroll
    for (int dy = 0; dy < 3; ++dy)
#pragma unroll
      for (int dx = 0; dx < 3; ++dx) {
        int hp = (iy + dy) * HALO + (ix + dx);
        const h16* kr = &KS[hp * CCH + cb];
        float d = 0.f;
#pragma unroll
        for (int j = 0; j < 16; ++j) d += qv[j] * (float)kr[j];
        d *= 0.25f;                    // 1/sqrt(DEPTH)
        sims[dy * 3 + dx] = d;
        mx = fmaxf(mx, d);
      }
    float se = 0.f;
#pragma unroll
    for (int i = 0; i < 9; ++i) { sims[i] = __expf(sims[i] - mx); se += sims[i]; }
    float inv = 1.f / se;

    float ov[16];
#pragma unroll
    for (int j = 0; j < 16; ++j) ov[j] = 0.f;
#pragma unroll
    for (int dy = 0; dy < 3; ++dy)
#pragma unroll
      for (int dx = 0; dx < 3; ++dx) {
        int hp = (iy + dy) * HALO + (ix + dx);
        const h16* vr = &VS[hp * CCH + cb];
        float a = sims[dy * 3 + dx] * inv;
#pragma unroll
        for (int j = 0; j < 16; ++j) ov[j] += a * (float)vr[j];
      }
    // residual with x (center halo pixel)
    int hpc = (iy + 1) * HALO + (ix + 1);
#pragma unroll
    for (int j = 0; j < 16; ++j) ov[j] += (float)XH[hpc * CCH + cb + j];

    // LayerNorm over 64 channels: partial sums + xor-shuffle across 4 lanes
    float s = 0.f, s2 = 0.f;
#pragma unroll
    for (int j = 0; j < 16; ++j) { s += ov[j]; s2 += ov[j] * ov[j]; }
    s  += __shfl_xor(s, 1);  s  += __shfl_xor(s, 2);
    s2 += __shfl_xor(s2, 1); s2 += __shfl_xor(s2, 2);
    float mean = s * (1.f / 64.f);
    float var  = s2 * (1.f / 64.f) - mean * mean;
    float rsd  = rsqrtf(var + LN_EPS);

    __syncthreads();                   // all LDS reads done before O1 overlay
#pragma unroll
    for (int j = 0; j < 16; ++j) {
      int c = cb + j;
      O1[p * CCH + c] = (h16)((ov[j] - mean) * rsd * ln1_g[c] + ln1_b[c]);
    }
  }
  __syncthreads();

  // ===== Phase 3: FFN layer 1 (64x64 @ 64x128) + bias + ReLU -> T1 ==========
  // 8 output columns of 16 -> exactly one per wave; B loaded once, 4 Mtiles.
  {
    int n = wv_id * 16 + lm;
    v16h b0 = pack_b_glb(ff1_w, FFD, n, 0, kg);
    v16h b1 = pack_b_glb(ff1_w, FFD, n, 32, kg);
    float bias = ff1_b[n];
#pragma unroll
    for (int mt = 0; mt < 4; ++mt) {
      const h16* arow = &O1[(mt * 16 + lm) * CCH];
      v8f c = {};
      c = wmma_f16(pack_a_lds(arow, 0, kg), b0, c);
      c = wmma_f16(pack_a_lds(arow, 32, kg), b1, c);
#pragma unroll
      for (int r = 0; r < 8; ++r)
        T1[(mt * 16 + r + (kg << 3)) * FFD + n] = (h16)fmaxf(c[r] + bias, 0.f);
    }
  }
  __syncthreads();

  // ===== Phase 4: FFN layer 2 (64x128 @ 128x64) + bias + residual -> O2 =====
  // wave w: column nt = w&3, Mtiles {2*(w>>2), 2*(w>>2)+1}; B loaded once.
  {
    int nt = wv_id & 3;
    int n  = nt * 16 + lm;
    v16h bf[4];
#pragma unroll
    for (int kk = 0; kk < 4; ++kk)
      bf[kk] = pack_b_glb(ff2_w, CCH, n, kk * 32, kg);
    float bias = ff2_b[n];
#pragma unroll
    for (int mi = 0; mi < 2; ++mi) {
      int mt = (wv_id >> 2) * 2 + mi;
      const h16* arow = &T1[(mt * 16 + lm) * FFD];
      v8f c = {};
#pragma unroll
      for (int kk = 0; kk < 4; ++kk)
        c = wmma_f16(pack_a_lds(arow, kk * 32, kg), bf[kk], c);
#pragma unroll
      for (int r = 0; r < 8; ++r) {
        int m = mt * 16 + r + (kg << 3);
        O2[m * CCH + n] = (h16)(c[r] + bias + (float)O1[m * CCH + n]);
      }
    }
  }
  __syncthreads();

  // ===== Phase 4b: LayerNorm2 in place on O2 ================================
  {
    int p = tid >> 2, qt = tid & 3, cb = qt << 4;
    float v[16];
#pragma unroll
    for (int j = 0; j < 16; ++j) v[j] = (float)O2[p * CCH + cb + j];
    float s = 0.f, s2 = 0.f;
#pragma unroll
    for (int j = 0; j < 16; ++j) { s += v[j]; s2 += v[j] * v[j]; }
    s  += __shfl_xor(s, 1);  s  += __shfl_xor(s, 2);
    s2 += __shfl_xor(s2, 1); s2 += __shfl_xor(s2, 2);
    float mean = s * (1.f / 64.f);
    float var  = s2 * (1.f / 64.f) - mean * mean;
    float rsd  = rsqrtf(var + LN_EPS);
#pragma unroll
    for (int j = 0; j < 16; ++j) {
      int c = cb + j;
      O2[p * CCH + c] = (h16)((v[j] - mean) * rsd * ln2_g[c] + ln2_b[c]);
    }
  }
  __syncthreads();

  // ===== Phase 5: fuse 1x1 conv (64x64 @ 64x64) + BN + ReLU -> global =======
  {
    int nt = wv_id & 3;
    int n  = nt * 16 + lm;
    v16h b0 = pack_b_glb(w_fuse, CCH, n, 0, kg);
    v16h b1 = pack_b_glb(w_fuse, CCH, n, 32, kg);
    float scale = bn_g[n] * rsqrtf(bn_var[n] + BN_EPS);
    float shift = bn_b[n] - bn_mean[n] * scale;
#pragma unroll
    for (int mi = 0; mi < 2; ++mi) {
      int mt = (wv_id >> 2) * 2 + mi;
      const h16* arow = &O2[(mt * 16 + lm) * CCH];
      v8f c = {};
      c = wmma_f16(pack_a_lds(arow, 0, kg), b0, c);
      c = wmma_f16(pack_a_lds(arow, 32, kg), b1, c);
#pragma unroll
      for (int r = 0; r < 8; ++r) {
        int p  = mt * 16 + r + (kg << 3);          // interior pixel 0..63
        int py = ty0 + (p >> 3);
        int px = tx0 + (p & 7);
        out[(((size_t)b * IMG + py) * IMG + px) * CCH + n] =
            fmaxf(c[r] * scale + shift, 0.f);
      }
    }
  }
}

extern "C" void kernel_launch(void* const* d_in, const int* in_sizes, int n_in,
                              void* d_out, int out_size, void* d_ws, size_t ws_size,
                              hipStream_t stream) {
  (void)in_sizes; (void)n_in; (void)out_size; (void)d_ws; (void)ws_size;
  const float* x      = (const float*)d_in[0];
  const float* wq     = (const float*)d_in[1];
  const float* wk     = (const float*)d_in[2];
  const float* wvv    = (const float*)d_in[3];
  const float* ln1_g  = (const float*)d_in[4];
  const float* ln1_b  = (const float*)d_in[5];
  const float* ff1_w  = (const float*)d_in[6];
  const float* ff1_b  = (const float*)d_in[7];
  const float* ff2_w  = (const float*)d_in[8];
  const float* ff2_b  = (const float*)d_in[9];
  const float* ln2_g  = (const float*)d_in[10];
  const float* ln2_b  = (const float*)d_in[11];
  const float* w_fuse = (const float*)d_in[12];
  const float* bn_g   = (const float*)d_in[13];
  const float* bn_b   = (const float*)d_in[14];
  const float* bn_m   = (const float*)d_in[15];
  const float* bn_v   = (const float*)d_in[16];
  float* out = (float*)d_out;

  dim3 grid(IMG / TILE, IMG / TILE, BATCH);   // 48 x 48 x 4
  dim3 block(256);                            // 8 waves (wave32)
  mhca_fused_kernel<<<grid, block, 0, stream>>>(
      x, wq, wk, wvv, ln1_g, ln1_b, ff1_w, ff1_b, ff2_w, ff2_b,
      ln2_g, ln2_b, w_fuse, bn_g, bn_b, bn_m, bn_v, out);
}
